// GRU_4269197492287
// MI455X (gfx1250) — matrix operand order repaired and checked by reading, not compiled, not run on hardware
//
#include <hip/hip_runtime.h>
#include <math.h>

// ---------------------------------------------------------------------------
// Types for CDNA5 WMMA (wave32): v_wmma_f32_16x16x32_bf16
// ---------------------------------------------------------------------------
typedef __bf16 bf16_t;
typedef __attribute__((ext_vector_type(8)))  bf16_t v8bf;
typedef __attribute__((ext_vector_type(16))) bf16_t v16bf;
typedef __attribute__((ext_vector_type(8)))  float  v8f;

#define WMMA_BF16(A_, B_, C_) \
  __builtin_amdgcn_wmma_f32_16x16x32_bf16(false, (A_), false, (B_), (short)0, (C_), false, false)

static constexpr int kT  = 128;   // time steps
static constexpr int kB  = 64;    // batch
static constexpr int kH  = 256;   // hidden
static constexpr int kH2 = 512;   // 2*hidden

// ---------------------------------------------------------------------------
// WMMA tile loaders (doc layouts, CDNA5 ISA §7.12.2):
//  A 16x32 bf16 : lane l holds row (l&15); K = 8*(l>>4) + {0..7, 16..23}
//  B 32x16 bf16 : lane l holds col (l&15); K = 8*(l>>4) + {0..7, 16..23}
//  C/D 16x16 f32: vgpr r -> row 8*(l>>4)+r, col (l&15)
// Weights are pre-transposed to (N x K) so B loads are contiguous in K.
// All bf16 operand tiles load as two 16-byte b128 chunks per lane.
// ---------------------------------------------------------------------------
static __device__ __forceinline__ v16bf ldA_bf(const bf16_t* src, int ld, int row0, int k0) {
  const int l = threadIdx.x & 31;
  const bf16_t* q = src + (size_t)(row0 + (l & 15)) * ld + (k0 + ((l >> 4) << 3));
  v8bf lo = *(const v8bf*)q;
  v8bf hi = *(const v8bf*)(q + 16);
  return __builtin_shufflevector(lo, hi, 0,1,2,3,4,5,6,7,8,9,10,11,12,13,14,15);
}

static __device__ __forceinline__ v16bf ldB_t(const bf16_t* wt, int ldk, int n0, int k0) {
  const int l = threadIdx.x & 31;
  const bf16_t* q = wt + (size_t)(n0 + (l & 15)) * ldk + (k0 + ((l >> 4) << 3));
  v8bf lo = *(const v8bf*)q;
  v8bf hi = *(const v8bf*)(q + 16);
  return __builtin_shufflevector(lo, hi, 0,1,2,3,4,5,6,7,8,9,10,11,12,13,14,15);
}

static __device__ __forceinline__ float sigmoidf_(float x) { return 1.f / (1.f + __expf(-x)); }

// ---------------------------------------------------------------------------
// Prep kernels: weight transpose+convert (K x N f32 -> N x K bf16), embedding
// ---------------------------------------------------------------------------
__global__ void k_conv_t(const float* __restrict__ src, bf16_t* __restrict__ dst, int K, int N) {
  int i = blockIdx.x * blockDim.x + threadIdx.x;
  if (i >= K * N) return;
  int k = i / N, n = i - k * N;
  dst[(size_t)n * K + k] = (bf16_t)src[i];
}

__global__ void k_embed(const int* __restrict__ tokens, const float* __restrict__ We,
                        bf16_t* __restrict__ emb, int TB, int H) {
  int i = blockIdx.x * blockDim.x + threadIdx.x;
  if (i >= TB * H) return;
  int tb = i / H, h = i - tb * H;
  emb[i] = (bf16_t)We[(size_t)tokens[tb] * H + h];
}

// ---------------------------------------------------------------------------
// Encoder: persistent per-direction GRU scan. One block per direction,
// 512 threads (16 waves). h state in LDS as f32 (recurrence precision) plus
// a bf16 mirror used as the WMMA A operand (written once per step).
// ---------------------------------------------------------------------------
struct EncW {
  const bf16_t *WxT, *WhT, *WxhT, *WhhT;
  const float *b, *bh;
};

__global__ __launch_bounds__(512)
void k_encoder(const bf16_t* __restrict__ emb, bf16_t* __restrict__ ctx, EncW wf, EncW wb) {
  extern __shared__ char smem[];
  float*  hS  = (float*)smem;                            // 64*256 f32  [0,64K)
  bf16_t* hbS = (bf16_t*)(smem + 64 * 1024);             // 64*256 bf16 [64K,96K)
  bf16_t* hrS = (bf16_t*)(smem + 96 * 1024);             // 64*256 bf16 [96K,128K)
  float*  zS  = (float*)(smem + 128 * 1024);             // 64*256 f32  [128K,192K)
  const int dir = blockIdx.x;
  const EncW W = dir ? wb : wf;
  const int tid = threadIdx.x, w = tid >> 5, l = tid & 31;
  for (int i = tid; i < kB * kH; i += 512) { hS[i] = 0.f; hbS[i] = (bf16_t)0.f; }
  __syncthreads();
  for (int s = 0; s < kT; ++s) {
    const int t = dir ? (kT - 1 - s) : s;
    const bf16_t* x = emb + (size_t)t * kB * kH;
    // Phase 1: rz = sigmoid(x@Wx + h@Wh + b); stage r*h (bf16) and z (f32)
    for (int half = 0; half < 2; ++half) {
      const int ni = w + 16 * half;  // 0..31 over 512 cols
      v8f acc[4] = {};
      for (int k0 = 0; k0 < kH; k0 += 32) {
        v16bf bx = ldB_t(W.WxT, kH, ni * 16, k0);
        v16bf bh2 = ldB_t(W.WhT, kH, ni * 16, k0);
#pragma unroll
        for (int mi = 0; mi < 4; ++mi) {
          v16bf ax = ldA_bf(x, kH, mi * 16, k0);
          acc[mi] = WMMA_BF16(ax, bx, acc[mi]);
          v16bf ah = ldA_bf(hbS, kH, mi * 16, k0);
          acc[mi] = WMMA_BF16(ah, bh2, acc[mi]);
        }
      }
      const int col = ni * 16 + (l & 15);
      const float bias = W.b[col];
#pragma unroll
      for (int mi = 0; mi < 4; ++mi)
#pragma unroll
        for (int r = 0; r < 8; ++r) {
          const int row = mi * 16 + ((l >> 4) << 3) + r;
          const float sg = sigmoidf_(acc[mi][r] + bias);
          if (half == 0) hrS[row * kH + col] = (bf16_t)(sg * hS[row * kH + col]);
          else           zS[row * kH + (col - kH)] = sg;
        }
    }
    __syncthreads();
    // Phase 2: hc = tanh(x@Wxh + (r*h)@Whh + bh); h = z*h + (1-z)*hc
    {
      const int ni = w;  // 0..15 over 256 cols
      v8f acc[4] = {};
      for (int k0 = 0; k0 < kH; k0 += 32) {
        v16bf bx = ldB_t(W.WxhT, kH, ni * 16, k0);
        v16bf bh2 = ldB_t(W.WhhT, kH, ni * 16, k0);
#pragma unroll
        for (int mi = 0; mi < 4; ++mi) {
          v16bf ax = ldA_bf(x, kH, mi * 16, k0);
          acc[mi] = WMMA_BF16(ax, bx, acc[mi]);
          v16bf ah = ldA_bf(hrS, kH, mi * 16, k0);
          acc[mi] = WMMA_BF16(ah, bh2, acc[mi]);
        }
      }
      const int col = ni * 16 + (l & 15);
      const float bias = W.bh[col];
      bf16_t* crow = ctx + (size_t)t * kB * kH2 + dir * kH;
#pragma unroll
      for (int mi = 0; mi < 4; ++mi)
#pragma unroll
        for (int r = 0; r < 8; ++r) {
          const int row = mi * 16 + ((l >> 4) << 3) + r;
          const float hc = tanhf(acc[mi][r] + bias);
          const float z = zS[row * kH + col];
          const float hn = z * hS[row * kH + col] + (1.f - z) * hc;
          hS[row * kH + col] = hn;
          hbS[row * kH + col] = (bf16_t)hn;
          crow[(size_t)row * kH2 + col] = (bf16_t)hn;
        }
    }
    __syncthreads();
  }
}

// ---------------------------------------------------------------------------
// pctx = context @ Wa_c + ba  : (T*B x 512) GEMM, K=512. 128 blocks x 64 rows.
// ---------------------------------------------------------------------------
__global__ __launch_bounds__(512)
void k_pctx(const bf16_t* __restrict__ ctx, const bf16_t* __restrict__ WacT,
            const float* __restrict__ ba, bf16_t* __restrict__ pctx) {
  const int w = threadIdx.x >> 5, l = threadIdx.x & 31;
  const bf16_t* A = ctx + (size_t)blockIdx.x * 64 * kH2;
  bf16_t* O = pctx + (size_t)blockIdx.x * 64 * kH2;
  for (int half = 0; half < 2; ++half) {
    const int ni = w + 16 * half;
    v8f acc[4] = {};
    for (int k0 = 0; k0 < kH2; k0 += 32) {
      v16bf b = ldB_t(WacT, kH2, ni * 16, k0);
#pragma unroll
      for (int mi = 0; mi < 4; ++mi) {
        v16bf a = ldA_bf(A, kH2, mi * 16, k0);
        acc[mi] = WMMA_BF16(a, b, acc[mi]);
      }
    }
    const int col = ni * 16 + (l & 15);
    const float bias = ba[col];
#pragma unroll
    for (int mi = 0; mi < 4; ++mi)
#pragma unroll
      for (int r = 0; r < 8; ++r) {
        const int row = mi * 16 + ((l >> 4) << 3) + r;
        O[(size_t)row * kH2 + col] = (bf16_t)(acc[mi][r] + bias);
      }
  }
}

// ---------------------------------------------------------------------------
// Decoder: persistent single block, 512 threads. Per step:
//  P1 g=h@Wa_h -> LDS   P2 scores(t,b)=tanh(pctx+g)·Wa_y   P3 softmax over t
//  P4 wc=Σ alphas*ctx   P5 rz GRU gates   P6 hc + h update
//  P7 lfc=wc@Wf_c       P8 fw gate, hf    P9 y=hf@Wy+by -> out
// LDS (288KB): h f32 | h bf16 mirror | bufA{g/wc/hf} | bufB{hr,z / lfc f32}
//              | scores | lfc bf16 mirror
// ---------------------------------------------------------------------------
__global__ __launch_bounds__(512)
void k_decoder(const bf16_t* __restrict__ ctx, const bf16_t* __restrict__ pctx,
               const bf16_t* WxdT, const bf16_t* WhdT, const float* b_d,
               const bf16_t* WxhdT, const bf16_t* WhhdT, const float* bh_d,
               const bf16_t* WahT, const float* Wa_y,
               const bf16_t* WfhT, const bf16_t* WfcT, const bf16_t* WffT,
               const float* bf_b, const float* Wy, const float* by,
               float* __restrict__ out) {
  extern __shared__ char smem[];
  float*  hS   = (float*)smem;                      // 64*256 f32   [0,64K)
  bf16_t* hbS  = (bf16_t*)(smem + 64 * 1024);       // 64*256 bf16  [64K,96K)
  char*   bufA = smem + 96 * 1024;                  // 64KB         [96K,160K)
  char*   bufB = smem + 160 * 1024;                 // 64KB         [160K,224K)
  float*  scS  = (float*)(smem + 224 * 1024);       // 128*64 f32   [224K,256K)
  bf16_t* lfbS = (bf16_t*)(smem + 256 * 1024);      // 64*256 bf16  [256K,288K)
  bf16_t* gS   = (bf16_t*)bufA;                     // 64*512 bf16 (P1,P2)
  bf16_t* wcS  = (bf16_t*)bufA;                     // 64*512 bf16 (P4..P7)
  float*  hfS  = (float*)bufA;                      // 64*256 f32  (P8,P9)
  bf16_t* hrS  = (bf16_t*)bufB;                     // 64*256 bf16 (P5,P6)
  bf16_t* zSd  = (bf16_t*)(bufB + 32 * 1024);       // 64*256 bf16 (P5,P6)
  float*  lfcS = (float*)bufB;                      // 64*256 f32  (P7,P8)
  const int tid = threadIdx.x, w = tid >> 5, l = tid & 31;
  for (int i = tid; i < kB * kH; i += 512) { hS[i] = 0.f; hbS[i] = (bf16_t)0.f; }
  __syncthreads();
  for (int s = 0; s < kT - 1; ++s) {
    const int t = s + 1;
    const bf16_t* x = ctx + (size_t)t * kB * kH2;
    // P1: g = h @ Wa_h
    for (int half = 0; half < 2; ++half) {
      const int ni = w + 16 * half;
      v8f acc[4] = {};
      for (int k0 = 0; k0 < kH; k0 += 32) {
        v16bf b = ldB_t(WahT, kH, ni * 16, k0);
#pragma unroll
        for (int mi = 0; mi < 4; ++mi) {
          v16bf a = ldA_bf(hbS, kH, mi * 16, k0);
          acc[mi] = WMMA_BF16(a, b, acc[mi]);
        }
      }
      const int col = ni * 16 + (l & 15);
#pragma unroll
      for (int mi = 0; mi < 4; ++mi)
#pragma unroll
        for (int r = 0; r < 8; ++r) {
          const int row = mi * 16 + ((l >> 4) << 3) + r;
          gS[row * kH2 + col] = (bf16_t)acc[mi][r];
        }
    }
    __syncthreads();
    // P2: scores[t][b] = sum_d tanh(pctx + g) * Wa_y   (vectorized b128 reads)
    for (int p = tid; p < kT * kB; p += 512) {
      const int tt = p >> 6, b = p & 63;
      const v8bf* pr = (const v8bf*)(pctx + ((size_t)tt * kB + b) * kH2);
      const v8bf* gr = (const v8bf*)(gS + b * kH2);
      float sacc = 0.f;
      for (int d8 = 0; d8 < kH2 / 8; ++d8) {
        v8bf pv = pr[d8];
        v8bf gv = gr[d8];
#pragma unroll
        for (int e = 0; e < 8; ++e)
          sacc += tanhf((float)pv[e] + (float)gv[e]) * Wa_y[d8 * 8 + e];
      }
      scS[tt * kB + b] = sacc;
    }
    __syncthreads();
    // P3: softmax over time per batch column
    if (tid < kB) {
      const int b = tid;
      float m = -3.0e38f;
      for (int tt = 0; tt < kT; ++tt) m = fmaxf(m, scS[tt * kB + b]);
      float ssum = 0.f;
      for (int tt = 0; tt < kT; ++tt) {
        float e = __expf(scS[tt * kB + b] - m);
        scS[tt * kB + b] = e; ssum += e;
      }
      const float inv = 1.f / ssum;
      for (int tt = 0; tt < kT; ++tt) scS[tt * kB + b] *= inv;
    }
    __syncthreads();
    // P4: wc[b][d] = sum_t alpha * ctx
    for (int o = tid; o < kB * kH2; o += 512) {
      const int b = o >> 9, d = o & (kH2 - 1);
      float a = 0.f;
      for (int tt = 0; tt < kT; ++tt)
        a += scS[tt * kB + b] * (float)ctx[((size_t)tt * kB + b) * kH2 + d];
      wcS[o] = (bf16_t)a;
    }
    __syncthreads();
    // P5: rz = sigmoid(x@Wx_d + h@Wh_d + b_d)
    for (int half = 0; half < 2; ++half) {
      const int ni = w + 16 * half;
      v8f acc[4] = {};
      for (int k0 = 0; k0 < kH2; k0 += 32) {
        v16bf b = ldB_t(WxdT, kH2, ni * 16, k0);
#pragma unroll
        for (int mi = 0; mi < 4; ++mi) {
          v16bf a = ldA_bf(x, kH2, mi * 16, k0);
          acc[mi] = WMMA_BF16(a, b, acc[mi]);
        }
      }
      for (int k0 = 0; k0 < kH; k0 += 32) {
        v16bf b = ldB_t(WhdT, kH, ni * 16, k0);
#pragma unroll
        for (int mi = 0; mi < 4; ++mi) {
          v16bf a = ldA_bf(hbS, kH, mi * 16, k0);
          acc[mi] = WMMA_BF16(a, b, acc[mi]);
        }
      }
      const int col = ni * 16 + (l & 15);
      const float bias = b_d[col];
#pragma unroll
      for (int mi = 0; mi < 4; ++mi)
#pragma unroll
        for (int r = 0; r < 8; ++r) {
          const int row = mi * 16 + ((l >> 4) << 3) + r;
          const float sg = sigmoidf_(acc[mi][r] + bias);
          if (half == 0) hrS[row * kH + col] = (bf16_t)(sg * hS[row * kH + col]);
          else           zSd[row * kH + (col - kH)] = (bf16_t)sg;
        }
    }
    __syncthreads();
    // P6: hc = tanh(x@Wxh_d + hr@Whh_d + bh_d); h = z*h + (1-z)*hc
    {
      const int ni = w;
      v8f acc[4] = {};
      for (int k0 = 0; k0 < kH2; k0 += 32) {
        v16bf b = ldB_t(WxhdT, kH2, ni * 16, k0);
#pragma unroll
        for (int mi = 0; mi < 4; ++mi) {
          v16bf a = ldA_bf(x, kH2, mi * 16, k0);
          acc[mi] = WMMA_BF16(a, b, acc[mi]);
        }
      }
      for (int k0 = 0; k0 < kH; k0 += 32) {
        v16bf b = ldB_t(WhhdT, kH, ni * 16, k0);
#pragma unroll
        for (int mi = 0; mi < 4; ++mi) {
          v16bf a = ldA_bf(hrS, kH, mi * 16, k0);
          acc[mi] = WMMA_BF16(a, b, acc[mi]);
        }
      }
      const int col = ni * 16 + (l & 15);
      const float bias = bh_d[col];
#pragma unroll
      for (int mi = 0; mi < 4; ++mi)
#pragma unroll
        for (int r = 0; r < 8; ++r) {
          const int row = mi * 16 + ((l >> 4) << 3) + r;
          const float hc = tanhf(acc[mi][r] + bias);
          const float z = (float)zSd[row * kH + col];
          const float hn = z * hS[row * kH + col] + (1.f - z) * hc;
          hS[row * kH + col] = hn;
          hbS[row * kH + col] = (bf16_t)hn;
        }
    }
    __syncthreads();
    // P7: lfc = wc @ Wf_c  (f32 into bufB + bf16 mirror; hr/z consumed)
    {
      const int ni = w;
      v8f acc[4] = {};
      for (int k0 = 0; k0 < kH2; k0 += 32) {
        v16bf b = ldB_t(WfcT, kH2, ni * 16, k0);
#pragma unroll
        for (int mi = 0; mi < 4; ++mi) {
          v16bf a = ldA_bf(wcS, kH2, mi * 16, k0);
          acc[mi] = WMMA_BF16(a, b, acc[mi]);
        }
      }
      const int col = ni * 16 + (l & 15);
#pragma unroll
      for (int mi = 0; mi < 4; ++mi)
#pragma unroll
        for (int r = 0; r < 8; ++r) {
          const int row = mi * 16 + ((l >> 4) << 3) + r;
          lfcS[row * kH + col] = acc[mi][r];
          lfbS[row * kH + col] = (bf16_t)acc[mi][r];
        }
    }
    __syncthreads();
    // P8: fw = sigmoid(lfc@Wf_f + h@Wf_h + bf); hf = lfc*fw + h (into bufA)
    {
      const int ni = w;
      v8f acc[4] = {};
      for (int k0 = 0; k0 < kH; k0 += 32) {
        v16bf b1 = ldB_t(WffT, kH, ni * 16, k0);
        v16bf b2 = ldB_t(WfhT, kH, ni * 16, k0);
#pragma unroll
        for (int mi = 0; mi < 4; ++mi) {
          v16bf a1 = ldA_bf(lfbS, kH, mi * 16, k0);
          acc[mi] = WMMA_BF16(a1, b1, acc[mi]);
          v16bf a2 = ldA_bf(hbS, kH, mi * 16, k0);
          acc[mi] = WMMA_BF16(a2, b2, acc[mi]);
        }
      }
      const int col = ni * 16 + (l & 15);
      const float bias = bf_b[col];
#pragma unroll
      for (int mi = 0; mi < 4; ++mi)
#pragma unroll
        for (int r = 0; r < 8; ++r) {
          const int row = mi * 16 + ((l >> 4) << 3) + r;
          const float fw = sigmoidf_(acc[mi][r] + bias);
          hfS[row * kH + col] = lfcS[row * kH + col] * fw + hS[row * kH + col];
        }
    }
    __syncthreads();
    // P9: y = hf @ Wy + by  (64 x 12)
    for (int o = tid; o < kB * 12; o += 512) {
      const int b = o / 12, v = o - b * 12;
      float a = by[v];
      for (int d = 0; d < kH; ++d) a += hfS[b * kH + d] * Wy[d * 12 + v];
      out[((size_t)s * kB + b) * 12 + v] = a;
    }
    __syncthreads();
  }
}

// ---------------------------------------------------------------------------
// Host launcher
// ---------------------------------------------------------------------------
extern "C" void kernel_launch(void* const* d_in, const int* in_sizes, int n_in,
                              void* d_out, int out_size, void* d_ws, size_t ws_size,
                              hipStream_t stream) {
  (void)in_sizes; (void)n_in; (void)out_size; (void)ws_size;
  char* ws = (char*)d_ws;
  size_t off = 0;
  auto alloc = [&](size_t bytes) -> char* {
    char* p = ws + off;
    off += (bytes + 255) & ~(size_t)255;
    return p;
  };
  bf16_t* emb  = (bf16_t*)alloc((size_t)kT * kB * kH * 2);   // 4 MB
  bf16_t* ctx  = (bf16_t*)alloc((size_t)kT * kB * kH2 * 2);  // 8 MB
  bf16_t* pctx = (bf16_t*)alloc((size_t)kT * kB * kH2 * 2);  // 8 MB

  auto wt = [&](int idx, int K, int N) -> bf16_t* {
    bf16_t* dst = (bf16_t*)alloc((size_t)K * N * 2);
    int n = K * N;
    k_conv_t<<<(n + 255) / 256, 256, 0, stream>>>((const float*)d_in[idx], dst, K, N);
    return dst;
  };
  // Encoder weights
  bf16_t* WxfT  = wt(2, kH, kH2);
  bf16_t* WhfT  = wt(3, kH, kH2);
  bf16_t* WxhfT = wt(5, kH, kH);
  bf16_t* WhhfT = wt(6, kH, kH);
  bf16_t* WxbT  = wt(8, kH, kH2);
  bf16_t* WhbT  = wt(9, kH, kH2);
  bf16_t* WxhbT = wt(11, kH, kH);
  bf16_t* WhhbT = wt(12, kH, kH);
  // Decoder GRU weights
  bf16_t* WxdT  = wt(14, kH2, kH2);
  bf16_t* WhdT  = wt(15, kH, kH2);
  bf16_t* WxhdT = wt(17, kH2, kH);
  bf16_t* WhhdT = wt(18, kH, kH);
  // Attention + fusion weights
  bf16_t* WahT  = wt(22, kH, kH2);
  bf16_t* WacT  = wt(23, kH2, kH2);
  bf16_t* WfhT  = wt(26, kH, kH);
  bf16_t* WfcT  = wt(27, kH2, kH);
  bf16_t* WffT  = wt(28, kH, kH);

  // Embedding gather -> bf16
  {
    int n = kT * kB * kH;
    k_embed<<<(n + 255) / 256, 256, 0, stream>>>((const int*)d_in[0], (const float*)d_in[1],
                                                 emb, kT * kB, kH);
  }

  // Bidirectional encoder: 2 blocks (one per direction), 192 KB LDS each
  EncW wf{WxfT, WhfT, WxhfT, WhhfT, (const float*)d_in[4], (const float*)d_in[7]};
  EncW wb{WxbT, WhbT, WxhbT, WhhbT, (const float*)d_in[10], (const float*)d_in[13]};
  k_encoder<<<2, 512, 192 * 1024, stream>>>(emb, ctx, wf, wb);

  // pctx GEMM: 128 blocks x 64 rows
  k_pctx<<<(kT * kB) / 64, 512, 0, stream>>>(ctx, WacT, (const float*)d_in[24], pctx);

  // Attention decoder: single persistent block, 288 KB LDS
  k_decoder<<<1, 512, 288 * 1024, stream>>>(
      ctx, pctx,
      WxdT, WhdT, (const float*)d_in[16],
      WxhdT, WhhdT, (const float*)d_in[19],
      WahT, (const float*)d_in[25],
      WfhT, WfcT, WffT,
      (const float*)d_in[29], (const float*)d_in[20], (const float*)d_in[21],
      (float*)d_out);
}
